// MultiHeadAttention_11519102288664
// MI455X (gfx1250) — matrix operand order repaired and checked by reading, not compiled, not run on hardware
//
#include <hip/hip_runtime.h>
#include <hip/hip_bf16.h>

typedef __attribute__((ext_vector_type(16))) _Float16 v16h;
typedef __attribute__((ext_vector_type(8)))  float    v8f;

constexpr int Bn  = 4;
constexpr int Sn  = 2048;
constexpr int Hn  = 1024;
constexpr int NHn = 16;
constexpr int HDn = 64;
constexpr int Mn  = Bn * Sn;           // 8192 rows for projection GEMMs

#define DEV static __device__ __forceinline__

// ---- WMMA fragment helpers (layouts per CDNA5 ISA 7.12.2) ----------------

// A-matrix 16x32 f16: lane l (M=l%16), hi = l/16 selects K-groups.
// VGPR0..3: K = hi*8 + {0,1},{2,3},{4,5},{6,7}; VGPR4..7: K = 16 + hi*8 + ...
DEV v16h frag_a(const _Float16* As, int lda, int row0, int koff, int l, int hi) {
  union { v16h v; unsigned u[8]; } r;
  const _Float16* p = As + (size_t)(row0 + l) * lda + koff;
#pragma unroll
  for (int i = 0; i < 8; ++i) {
    int kb = (i < 4) ? (hi * 8 + 2 * i) : (16 + hi * 8 + 2 * (i - 4));
    r.u[i] = *(const unsigned*)(p + kb);
  }
  return r.v;
}

// B-matrix 32x16 f16: N = lane%16, K order matches frag_a: K = hi*8 + 0..7, then 16 + hi*8 + 0..7.
// Bt is stored N-major in LDS (Bt[n][k]) so each lane's Ks are contiguous.
DEV v16h frag_b(const _Float16* Bt, int ldb, int col0, int koff, int l, int hi) {
  union { v16h v; unsigned u[8]; } r;
  const _Float16* p = Bt + (size_t)(col0 + l) * ldb + koff;
#pragma unroll
  for (int i = 0; i < 8; ++i) {
    int kb = (i < 4) ? (hi * 8 + 2 * i) : (16 + hi * 8 + 2 * (i - 4));
    r.u[i] = *(const unsigned*)(p + kb);
  }
  return r.v;
}

DEV v8f wmma16(v16h a, v16h b, v8f c) {
  return __builtin_amdgcn_wmma_f32_16x16x32_f16(
      /*neg_a=*/false, a, /*neg_b=*/false, b,
      /*c_mod=*/(short)0, c, /*reuse_a=*/false, /*reuse_b=*/false);
}

// Async global->LDS DMA: copies 16 bytes per lane, tracked by ASYNCcnt.
// lds_off = LDS byte offset (low 32 bits of the generic shared address).
DEV void async_copy_b128(unsigned lds_off, const void* gptr) {
  asm volatile("global_load_async_to_lds_b128 %0, %1, off"
               :: "v"(lds_off), "v"(gptr)
               : "memory");
}

DEV void wait_async0() {
  asm volatile("s_wait_asynccnt 0x0" ::: "memory");
}

// ---- Kernel 1/5: projection GEMM  out[M,N] = X[M,K] * W[K,N] + bias -------
// Block: 256 threads (8 waves), 64x64 tile; each wave computes 16x32 (2 WMMAs).
template <bool A_F16, bool SPLIT>
__global__ __launch_bounds__(256) void gemm_proj(const void* __restrict__ Xv,
                                                 const float* __restrict__ W,
                                                 const float* __restrict__ bias,
                                                 void* __restrict__ outv) {
  __shared__ _Float16 As[64 * 34];
  __shared__ _Float16 Bs[64 * 34];
  const int tid = threadIdx.x;
  const int lane = tid & 31, w = tid >> 5;
  const int l = lane & 15, hi = lane >> 4;
  const int n0 = blockIdx.x * 64, m0 = blockIdx.y * 64;
  const int mw = (w & 3) * 16, nw = (w >> 2) * 32;
  const float*    Xf = (const float*)Xv;
  const _Float16* Xh = (const _Float16*)Xv;

  v8f acc0 = {}, acc1 = {};
  for (int k0 = 0; k0 < Hn; k0 += 32) {
    // A tile 64x32 (convert to f16)
#pragma unroll
    for (int j = 0; j < 8; ++j) {
      int i = tid + j * 256;
      int r = i >> 5, k = i & 31;
      float x = A_F16 ? (float)Xh[(size_t)(m0 + r) * Hn + k0 + k]
                      : Xf[(size_t)(m0 + r) * Hn + k0 + k];
      As[r * 34 + k] = (_Float16)x;
    }
    // B tile 32x64, stored transposed: Bs[n][k] = W[k0+k][n0+n]
#pragma unroll
    for (int j = 0; j < 8; ++j) {
      int i = tid + j * 256;
      int k = i >> 6, n = i & 63;
      Bs[n * 34 + k] = (_Float16)W[(size_t)(k0 + k) * Hn + n0 + n];
    }
    __syncthreads();
    v16h a  = frag_a(As, 34, mw, 0, l, hi);
    v16h b0 = frag_b(Bs, 34, nw, 0, l, hi);
    v16h b1 = frag_b(Bs, 34, nw + 16, 0, l, hi);
    acc0 = wmma16(a, b0, acc0);
    acc1 = wmma16(a, b1, acc1);
    __syncthreads();
  }
  // D layout: VGPR v, lane -> M = v + 8*hi, N = l
#pragma unroll
  for (int t = 0; t < 2; ++t) {
    v8f a = t ? acc1 : acc0;
    int gc = n0 + nw + t * 16 + l;
    float bv = bias[gc];
#pragma unroll
    for (int v = 0; v < 8; ++v) {
      int gr = m0 + mw + 8 * hi + v;
      float val = a[v] + bv;
      if (SPLIT) {  // f16 out, [B,NH,S,HD]
        int b = gr >> 11, s = gr & (Sn - 1);
        int head = gc >> 6, d = gc & 63;
        ((_Float16*)outv)[((size_t)(b * NHn + head) * Sn + s) * HDn + d] =
            (_Float16)val;
      } else {      // f32 out, [M,N]
        ((float*)outv)[(size_t)gr * Hn + gc] = val;
      }
    }
  }
}

// ---- Kernel 2: energy = (Q Kt)/sqrt(HD) with mask ------------------------
// Q/K tiles staged via async global->LDS DMA (pure f16 byte copy, no VGPR
// round-trip); stride 72 halves = 144 B keeps every b128 LDS write aligned.
__global__ __launch_bounds__(256) void attn_energy(
    const _Float16* __restrict__ Q, const _Float16* __restrict__ Kh,
    const int* __restrict__ mask, float* __restrict__ attn) {
  __shared__ _Float16 Qs[64 * 72];
  __shared__ _Float16 Ks[64 * 72];  // Bt layout: Ks[col][d] == K[col][d]
  const int tid = threadIdx.x, lane = tid & 31, w = tid >> 5;
  const int l = lane & 15, hi = lane >> 4;
  const int k0 = blockIdx.x * 64, q0 = blockIdx.y * 64, bh = blockIdx.z;
  const int b = bh >> 4;
  const _Float16* Qb = Q + (size_t)bh * Sn * HDn;
  const _Float16* Kb = Kh + (size_t)bh * Sn * HDn;

  const unsigned qbase = (unsigned)(size_t)(&Qs[0]);
  const unsigned kbase = (unsigned)(size_t)(&Ks[0]);
  // 64 rows x 128 B per tile = 512 16-byte chunks; 2 per thread per tile.
#pragma unroll
  for (int j = 0; j < 2; ++j) {
    int c = tid + j * 256;
    int r = c >> 3, cc = c & 7;
    async_copy_b128(qbase + r * 144 + cc * 16,
                    Qb + (size_t)(q0 + r) * HDn + cc * 8);
    async_copy_b128(kbase + r * 144 + cc * 16,
                    Kb + (size_t)(k0 + r) * HDn + cc * 8);
  }
  wait_async0();
  __syncthreads();

  const int mw = (w & 3) * 16, nw = (w >> 2) * 32;
  v8f acc0 = {}, acc1 = {};
#pragma unroll
  for (int ks = 0; ks < 2; ++ks) {  // K-dim = HD = 64 = 2 wmma steps
    v16h a  = frag_a(Qs, 72, mw, ks * 32, l, hi);
    v16h b0 = frag_b(Ks, 72, nw, ks * 32, l, hi);
    v16h b1 = frag_b(Ks, 72, nw + 16, ks * 32, l, hi);
    acc0 = wmma16(a, b0, acc0);
    acc1 = wmma16(a, b1, acc1);
  }
#pragma unroll
  for (int t = 0; t < 2; ++t) {
    v8f a = t ? acc1 : acc0;
    int gc = k0 + nw + t * 16 + l;
    bool keep = mask[b * Sn + gc] != 0;
#pragma unroll
    for (int v = 0; v < 8; ++v) {
      int gr = q0 + mw + 8 * hi + v;
      float val = a[v] * 0.125f;  // 1/sqrt(64)
      if (!keep) val = -1e10f;
      attn[(size_t)(bh * Sn + gr) * Sn + gc] = val;
    }
  }
}

// ---- Kernel 3: row softmax over 2048 columns, in place -------------------
__global__ __launch_bounds__(256) void softmax_rows(float* __restrict__ attn) {
  __shared__ float red[256];
  const int tid = threadIdx.x;
  float* p = attn + (size_t)blockIdx.x * Sn;
  float v[8];
#pragma unroll
  for (int j = 0; j < 8; ++j) v[j] = p[tid + j * 256];
  float m = v[0];
#pragma unroll
  for (int j = 1; j < 8; ++j) m = fmaxf(m, v[j]);
  red[tid] = m;
  __syncthreads();
  for (int s = 128; s > 0; s >>= 1) {
    if (tid < s) red[tid] = fmaxf(red[tid], red[tid + s]);
    __syncthreads();
  }
  float rowmax = red[0];
  __syncthreads();
  float e[8], sum = 0.f;
#pragma unroll
  for (int j = 0; j < 8; ++j) {
    e[j] = __expf(v[j] - rowmax);
    sum += e[j];
  }
  red[tid] = sum;
  __syncthreads();
  for (int s = 128; s > 0; s >>= 1) {
    if (tid < s) red[tid] += red[tid + s];
    __syncthreads();
  }
  float inv = 1.0f / red[0];
#pragma unroll
  for (int j = 0; j < 8; ++j) p[tid + j * 256] = e[j] * inv;
}

// ---- Kernel 4: X = P V  (per bh: [S,S] x [S,HD]) -------------------------
__global__ __launch_bounds__(256) void attn_pv(const float* __restrict__ attn,
                                               const _Float16* __restrict__ V,
                                               _Float16* __restrict__ X) {
  __shared__ _Float16 Ps[64 * 34];
  __shared__ _Float16 Vt[64 * 34];  // Bt layout: Vt[d][s]
  const int tid = threadIdx.x, lane = tid & 31, w = tid >> 5;
  const int l = lane & 15, hi = lane >> 4;
  const int q0 = blockIdx.x * 64, bh = blockIdx.y;
  const int b = bh >> 4, head = bh & 15;
  const float*    Pb = attn + (size_t)bh * Sn * Sn;
  const _Float16* Vb = V + (size_t)bh * Sn * HDn;
  const int mw = (w & 3) * 16, nw = (w >> 2) * 32;
  v8f acc0 = {}, acc1 = {};
  for (int k0 = 0; k0 < Sn; k0 += 32) {
#pragma unroll
    for (int j = 0; j < 8; ++j) {
      int i = tid + j * 256;
      int r = i >> 5, k = i & 31;
      Ps[r * 34 + k] = (_Float16)Pb[(size_t)(q0 + r) * Sn + k0 + k];
    }
#pragma unroll
    for (int j = 0; j < 8; ++j) {
      int i = tid + j * 256;
      int s = i >> 6, d = i & 63;
      Vt[d * 34 + s] = Vb[(size_t)(k0 + s) * HDn + d];
    }
    __syncthreads();
    v16h a  = frag_a(Ps, 34, mw, 0, l, hi);
    v16h b0 = frag_b(Vt, 34, nw, 0, l, hi);
    v16h b1 = frag_b(Vt, 34, nw + 16, 0, l, hi);
    acc0 = wmma16(a, b0, acc0);
    acc1 = wmma16(a, b1, acc1);
    __syncthreads();
  }
  // write X as [B,S,H] f16 (H = NH*HD), ready for output projection
#pragma unroll
  for (int t = 0; t < 2; ++t) {
    v8f a = t ? acc1 : acc0;
    int d = nw + t * 16 + l;
#pragma unroll
    for (int v = 0; v < 8; ++v) {
      int s = q0 + mw + 8 * hi + v;
      X[(size_t)(b * Sn + s) * Hn + head * HDn + d] = (_Float16)a[v];
    }
  }
}

// ---- Host launch ---------------------------------------------------------
extern "C" void kernel_launch(void* const* d_in, const int* in_sizes, int n_in,
                              void* d_out, int out_size, void* d_ws,
                              size_t ws_size, hipStream_t stream) {
  const float* query = (const float*)d_in[0];
  const float* key   = (const float*)d_in[1];
  const float* value = (const float*)d_in[2];
  const int*   mask  = (const int*)d_in[3];
  const float* Wq = (const float*)d_in[4];
  const float* bq = (const float*)d_in[5];
  const float* Wk = (const float*)d_in[6];
  const float* bk = (const float*)d_in[7];
  const float* Wv = (const float*)d_in[8];
  const float* bv = (const float*)d_in[9];
  const float* Wo = (const float*)d_in[10];
  const float* bo = (const float*)d_in[11];

  float* out  = (float*)d_out;
  float* attn = out + (size_t)Bn * Sn * Hn;  // attention output region

  const size_t QKV = (size_t)Bn * NHn * Sn * HDn;  // 8,388,608 halves
  _Float16* ws = (_Float16*)d_ws;
  _Float16* Qh = ws;
  _Float16* Kh = ws + QKV;
  _Float16* Vh = ws + 2 * QKV;
  _Float16* Xh = ws;  // reuse Q region after energy pass (48 MB total ws)

  dim3 blk(256);
  dim3 gproj(Hn / 64, Mn / 64);  // (16, 128)

  gemm_proj<false, true><<<gproj, blk, 0, stream>>>(query, Wq, bq, Qh);
  gemm_proj<false, true><<<gproj, blk, 0, stream>>>(key,   Wk, bk, Kh);
  gemm_proj<false, true><<<gproj, blk, 0, stream>>>(value, Wv, bv, Vh);

  attn_energy<<<dim3(Sn / 64, Sn / 64, Bn * NHn), blk, 0, stream>>>(Qh, Kh,
                                                                    mask, attn);
  softmax_rows<<<dim3(Bn * NHn * Sn), blk, 0, stream>>>(attn);
  attn_pv<<<dim3(Sn / 64, Bn * NHn), blk, 0, stream>>>(attn, Vh, Xh);

  gemm_proj<true, false><<<gproj, blk, 0, stream>>>(Xh, Wo, bo, out);
}
